// Attention_44684839748079
// MI455X (gfx1250) — matrix operand order repaired and checked by reading, not compiled, not run on hardware
//
#include <hip/hip_runtime.h>

typedef __bf16 bf16;
typedef unsigned int u32;
typedef __attribute__((ext_vector_type(16))) __bf16 bf16x16;
typedef __attribute__((ext_vector_type(8)))  __bf16 bf16x8;
typedef __attribute__((ext_vector_type(4)))  __bf16 bf16x4;
typedef __attribute__((ext_vector_type(8)))  float  f32x8;
typedef __attribute__((ext_vector_type(4)))  u32    u32x4;
typedef __attribute__((ext_vector_type(8)))  int    i32x8;
typedef __attribute__((ext_vector_type(4)))  int    i32x4;

// ---------------------------------------------------------------------------
// WMMA helpers (CDNA5 gfx1250, wave32, 16x16x32 bf16 -> f32)
// ---------------------------------------------------------------------------
static __device__ __forceinline__ f32x8 wmma_bf16(bf16x16 a, bf16x16 b, f32x8 c) {
  return __builtin_amdgcn_wmma_f32_16x16x32_bf16(false, a, false, b, (short)0, c,
                                                 false, false);
}

// A fragment: 16x32 (MxK) tile, row-major with leading dim ld.
static __device__ __forceinline__ bf16x16 load_a_frag(const bf16* __restrict__ base,
                                                      int ld, int lane) {
  const int m  = lane & 15;
  const int hi = (lane >> 4) & 1;
  const bf16* row = base + (size_t)m * ld + hi * 16;
  bf16x8 lo = *(const bf16x8*)(row);
  bf16x8 hh = *(const bf16x8*)(row + 8);
  bf16x16 a;
#pragma unroll
  for (int i = 0; i < 8; ++i) { a[i] = lo[i]; a[i + 8] = hh[i]; }
  return a;
}

// B fragment: 32x16 (KxN), element(k,n) = base[n*ld + k] (W stored [N,K] row-major).
static __device__ __forceinline__ bf16x16 load_b_frag(const bf16* __restrict__ base,
                                                      int ld, int lane) {
  const int n  = lane & 15;
  const int hi = (lane >> 4) & 1;
  return *(const bf16x16*)(base + (size_t)n * ld + hi * 16);
}

// ---------------------------------------------------------------------------
// Tensor Data Mover: 2D tile (bf16) global -> LDS, row-major compact in LDS.
// Builds D# groups per CDNA5 ISA 8.3/8.4; clang-23 6-arg builtin form.
// ---------------------------------------------------------------------------
static __device__ __forceinline__ void tdm_load_2d_bf16(
    u32 lds_addr, const void* gptr, u32 tensor_d0, u32 tensor_d1,
    u32 tile_d0, u32 tile_d1, u32 row_stride_elems) {
  unsigned long long ga = (unsigned long long)(size_t)gptr;
  u32x4 g0;
  g0[0] = 1u;                                            // count=1, user mode
  g0[1] = lds_addr;                                      // LDS byte address
  g0[2] = (u32)ga;                                       // global addr lo
  g0[3] = (u32)((ga >> 32) & 0x01FFFFFFu) | (2u << 30);  // addr hi | type=2
  i32x8 g1;
  g1[0] = (int)(1u << 16);                               // data_size = 2 bytes
  g1[1] = (int)(tensor_d0 << 16);                        // tensor_dim0 lo16
  g1[2] = (int)((tensor_d0 >> 16) | (tensor_d1 << 16));  // dim0 hi / dim1 lo
  g1[3] = (int)((tensor_d1 >> 16) | (tile_d0 << 16));    // dim1 hi / tile_dim0
  g1[4] = (int)(tile_d1 & 0xFFFFu);                      // tile_dim1, tile_dim2=0
  g1[5] = (int)row_stride_elems;                         // tensor_dim0_stride lo32
  g1[6] = 0;                                             // stride0 hi / stride1 lo
  g1[7] = 0;
  i32x4 z4 = {0, 0, 0, 0};
  i32x8 z8 = {0, 0, 0, 0, 0, 0, 0, 0};
  __builtin_amdgcn_tensor_load_to_lds(g0, g1, z4, z4, z8, 0);
}

// ---------------------------------------------------------------------------
// fp32 -> bf16 conversion (x4 vectorized)
// ---------------------------------------------------------------------------
__global__ void cvt_f32_bf16(const float* __restrict__ in, bf16* __restrict__ out,
                             int n4) {
  int i = blockIdx.x * blockDim.x + threadIdx.x;
  if (i < n4) {
    const float4 v = ((const float4*)in)[i];
    bf16x4 o = {(bf16)v.x, (bf16)v.y, (bf16)v.z, (bf16)v.w};
    ((bf16x4*)out)[i] = o;
  }
}

// ---------------------------------------------------------------------------
// Y[M,N] = X[M,K] * W[N,K]^T. 4 waves, each a 64x64 tile (4x4 wmma frags),
// block tile 128x128. 16 WMMAs per k-step per wave.
// ---------------------------------------------------------------------------
template <bool OUT_F32>
__global__ __launch_bounds__(128) void gemm_xt_wmma(const bf16* __restrict__ A,
                                                    const bf16* __restrict__ W,
                                                    void* __restrict__ OutP,
                                                    int Mr, int Nr, int Kr) {
  const int lane = threadIdx.x & 31;
  const int wave = threadIdx.x >> 5;
  const int row0 = blockIdx.y * 128 + (wave >> 1) * 64;
  const int col0 = blockIdx.x * 128 + (wave & 1) * 64;

  f32x8 acc[4][4] = {};
  for (int k0 = 0; k0 < Kr; k0 += 32) {
    bf16x16 a[4], b[4];
#pragma unroll
    for (int i = 0; i < 4; ++i)
      a[i] = load_a_frag(A + (size_t)(row0 + 16 * i) * Kr + k0, Kr, lane);
#pragma unroll
    for (int j = 0; j < 4; ++j)
      b[j] = load_b_frag(W + (size_t)(col0 + 16 * j) * Kr + k0, Kr, lane);
#pragma unroll
    for (int i = 0; i < 4; ++i)
#pragma unroll
      for (int j = 0; j < 4; ++j)
        acc[i][j] = wmma_bf16(a[i], b[j], acc[i][j]);
  }

  const int nn = lane & 15, hi = (lane >> 4) & 1;
#pragma unroll
  for (int i = 0; i < 4; ++i)
#pragma unroll
    for (int j = 0; j < 4; ++j)
#pragma unroll
      for (int r = 0; r < 8; ++r) {
        size_t off = (size_t)(row0 + 16 * i + r + 8 * hi) * Nr +
                     (size_t)(col0 + 16 * j + nn);
        if (OUT_F32) ((float*)OutP)[off] = acc[i][j][r];
        else         ((bf16*)OutP)[off] = (bf16)acc[i][j][r];
      }
}

// ---------------------------------------------------------------------------
// RoPE in place on bf16 buffer [B*T, heads*64], rotate-half pairing (d, d+32).
// ---------------------------------------------------------------------------
__global__ void rope_kernel(bf16* __restrict__ q, const int* __restrict__ pos,
                            int rows, int heads, int T) {
  int idx = blockIdx.x * blockDim.x + threadIdx.x;
  int total = rows * heads * 32;
  if (idx >= total) return;
  int i   = idx & 31;
  int h   = (idx >> 5) % heads;
  int row = idx / (32 * heads);
  int t   = row % T;
  float p = (float)pos[t];
  const float lnt = 9.210340371976184f;  // ln(10000)
  int j = i >> 1;
  float f_lo = p * __expf(-((float)j / 32.f) * lnt);
  float f_hi = p * __expf(-((float)(j + 16) / 32.f) * lnt);
  bf16* base = q + (size_t)row * (heads * 64) + h * 64;
  float lo = (float)base[i];
  float hv = (float)base[i + 32];
  base[i]      = (bf16)(lo * __cosf(f_lo) - hv * __sinf(f_lo));
  base[i + 32] = (bf16)(hv * __cosf(f_hi) + lo * __sinf(f_hi));
}

// ---------------------------------------------------------------------------
// Flash attention: block = 4 waves, each wave owns a 16-row Q tile.
// 64-key blocks: K staged by TDM, V staged transposed (vectorized b128 LDS
// stores), row-max via LDS float atomics, row-sum via WMMA-with-ones.
// Per iteration per wave: 8 S-WMMAs + 2 sum-WMMAs + 8 PV-WMMAs.
// ---------------------------------------------------------------------------
__global__ __launch_bounds__(128) void flash_attn(const bf16* __restrict__ Q,
                                                  const bf16* __restrict__ Kb,
                                                  const bf16* __restrict__ Vb,
                                                  bf16* __restrict__ Y) {
  constexpr int T = 2048, Cq = 2048, Ckv = 512;
  __shared__ __align__(32) bf16 sK[64 * 64];        // K block row-major (TDM dest)
  __shared__ __align__(32) bf16 sVt[64 * 64];       // V^T: [d][key_local]
  __shared__ __align__(32) bf16 sP[4][16 * 64];     // per-wave P tile
  __shared__ __align__(16) float sMax[4][16];       // per-wave row-max slots

  const int lane = threadIdx.x & 31;
  const int wave = threadIdx.x >> 5;
  const int b = blockIdx.z, h = blockIdx.y;
  const int kvh = h >> 2;                            // GQA repeat_interleave (rep=4)
  const int q0 = blockIdx.x * 64 + wave * 16;
  const int nn = lane & 15, hi = (lane >> 4) & 1;

  const bf16* qbase = Q + (size_t)(b * T + q0) * Cq + h * 64;
  bf16x16 qf0 = load_a_frag(qbase, Cq, lane);
  bf16x16 qf1 = load_a_frag(qbase + 32, Cq, lane);

  bf16x16 ones;
#pragma unroll
  for (int i = 0; i < 16; ++i) ones[i] = (bf16)1.0f;

  f32x8 o[4] = {};
  float mrow[8], lrow[8];
#pragma unroll
  for (int r = 0; r < 8; ++r) { mrow[r] = -1e30f; lrow[r] = 0.f; }

  const int kEnd = blockIdx.x * 64 + 64;             // causal bound (uniform per block)
  for (int k0 = 0; k0 < kEnd; k0 += 64) {
    __syncthreads();                                 // protect sK/sVt reuse
    // ---- TDM: stage K block [64 keys x 64 d] row-major into sK ----
    if (wave == 0) {
      tdm_load_2d_bf16((u32)(size_t)(void*)sK,
                       Kb + (size_t)(b * T + k0) * Ckv + kvh * 64,
                       /*tensor_d0=*/64, /*tensor_d1=*/(u32)(T - k0),
                       /*tile_d0=*/64, /*tile_d1=*/64,
                       /*row_stride=*/Ckv);
      __builtin_amdgcn_s_wait_tensorcnt(0);
    }
    // ---- V staging: 64 keys x 64 d -> sVt[d][key], packed b128 stores ----
    {
      int kg = threadIdx.x >> 4;                     // key group: keys 8*kg..8*kg+7
      int dc = threadIdx.x & 15;                     // d chunk: d 4*dc..4*dc+3
      const bf16* vcol = Vb + (size_t)(b * T + k0 + 8 * kg) * Ckv + kvh * 64 + 4 * dc;
      bf16x4 vv[8];
#pragma unroll
      for (int kk = 0; kk < 8; ++kk)
        vv[kk] = *(const bf16x4*)(vcol + (size_t)kk * Ckv);
#pragma unroll
      for (int e = 0; e < 4; ++e) {
        bf16x8 pack;
#pragma unroll
        for (int kk = 0; kk < 8; ++kk) pack[kk] = vv[kk][e];
        *(bf16x8*)(&sVt[(4 * dc + e) * 64 + 8 * kg]) = pack;
      }
    }
    __syncthreads();

    // ---- S = Q K^T from LDS-resident K (four 16-key subtiles) ----
    f32x8 s[4];
#pragma unroll
    for (int jt = 0; jt < 4; ++jt) {
      const bf16* kbase = sK + (size_t)(jt * 16) * 64;
      f32x8 c = {};
      c = wmma_bf16(qf0, load_b_frag(kbase, 64, lane), c);
      c = wmma_bf16(qf1, load_b_frag(kbase + 32, 64, lane), c);
      s[jt] = c;
    }

    // ---- scale + causal mask ----
    const float scale = 0.125f;                      // 1/sqrt(64)
#pragma unroll
    for (int jt = 0; jt < 4; ++jt)
#pragma unroll
      for (int r = 0; r < 8; ++r) {
        int key  = k0 + jt * 16 + nn;
        int qrow = q0 + r + 8 * hi;
        float v = s[jt][r] * scale;
        s[jt][r] = (key <= qrow) ? v : -1e30f;
      }

    // ---- row max via per-wave LDS float atomics (in-order LDS pipe) ----
    sMax[wave][nn] = -3e30f;                         // init (dup lanes benign)
    float part[8];
#pragma unroll
    for (int r = 0; r < 8; ++r) {
      part[r] = fmaxf(fmaxf(s[0][r], s[1][r]), fmaxf(s[2][r], s[3][r]));
      __hip_atomic_fetch_max(&sMax[wave][r + 8 * hi], part[r],
                             __ATOMIC_RELAXED, __HIP_MEMORY_SCOPE_WORKGROUP);
    }
    asm volatile("s_wait_dscnt 0x0" ::: "memory");
    float4 ma = *(const float4*)(&sMax[wave][8 * hi]);
    float4 mb = *(const float4*)(&sMax[wave][8 * hi + 4]);
    float rmax[8] = {ma.x, ma.y, ma.z, ma.w, mb.x, mb.y, mb.z, mb.w};

    float alpha[8];
#pragma unroll
    for (int r = 0; r < 8; ++r) {
      float mn = fmaxf(mrow[r], rmax[r]);
      alpha[r] = __expf(mrow[r] - mn);
      mrow[r] = mn;
    }
#pragma unroll
    for (int jt = 0; jt < 4; ++jt)
#pragma unroll
      for (int r = 0; r < 8; ++r) s[jt][r] = __expf(s[jt][r] - mrow[r]);

    // ---- transpose P (D-layout) -> A-fragment layout via per-wave LDS ----
    bf16* pw = &sP[wave][0];
#pragma unroll
    for (int jt = 0; jt < 4; ++jt)
#pragma unroll
      for (int r = 0; r < 8; ++r)
        pw[(r + 8 * hi) * 64 + jt * 16 + nn] = (bf16)s[jt][r];
    asm volatile("s_wait_dscnt 0x0" ::: "memory");
    bf16x16 pf0 = load_a_frag(pw, 64, lane);         // keys 0..31
    bf16x16 pf1 = load_a_frag(pw + 32, 64, lane);    // keys 32..63

    // ---- row sums via WMMA with all-ones B (every D column = row sum) ----
    f32x8 zz = {};
    f32x8 rs = wmma_bf16(pf0, ones, zz);
    rs = wmma_bf16(pf1, ones, rs);
#pragma unroll
    for (int r = 0; r < 8; ++r) lrow[r] = lrow[r] * alpha[r] + rs[r];
#pragma unroll
    for (int nt = 0; nt < 4; ++nt)
#pragma unroll
      for (int r = 0; r < 8; ++r) o[nt][r] *= alpha[r];

    // ---- O += P * V ----
#pragma unroll
    for (int nt = 0; nt < 4; ++nt) {
      bf16x16 v0 = *(const bf16x16*)(&sVt[(nt * 16 + nn) * 64 + hi * 16]);
      bf16x16 v1 = *(const bf16x16*)(&sVt[(nt * 16 + nn) * 64 + 32 + hi * 16]);
      o[nt] = wmma_bf16(pf0, v0, o[nt]);
      o[nt] = wmma_bf16(pf1, v1, o[nt]);
    }
  }

  // ---- epilogue: normalize and write [b*T+t, h*64 + d] ----
  bf16* yb = Y + (size_t)(b * T + q0) * 2048 + h * 64;
  float rinv[8];
#pragma unroll
  for (int r = 0; r < 8; ++r) rinv[r] = __frcp_rn(lrow[r]);
#pragma unroll
  for (int nt = 0; nt < 4; ++nt)
#pragma unroll
    for (int r = 0; r < 8; ++r)
      yb[(size_t)(r + 8 * hi) * 2048 + nt * 16 + nn] = (bf16)(o[nt][r] * rinv[r]);
}

// ---------------------------------------------------------------------------
// Orchestration
// ---------------------------------------------------------------------------
extern "C" void kernel_launch(void* const* d_in, const int* in_sizes, int n_in,
                              void* d_out, int out_size, void* d_ws, size_t ws_size,
                              hipStream_t stream) {
  constexpr int B = 2, T = 2048, C = 2048, H = 32, HKV = 8, D = 64;
  constexpr size_t M = (size_t)B * T;

  const float* x  = (const float*)d_in[0];
  const float* Wq = (const float*)d_in[1];
  const float* Wk = (const float*)d_in[2];
  const float* Wv = (const float*)d_in[3];
  const float* Wc = (const float*)d_in[4];
  const int*  pos = (const int*)d_in[5];

  char* w = (char*)d_ws;
  bf16* xb  = (bf16*)w; w += M * C * 2;
  bf16* wqb = (bf16*)w; w += (size_t)H * D * C * 2;
  bf16* wkb = (bf16*)w; w += (size_t)HKV * D * C * 2;
  bf16* wvb = (bf16*)w; w += (size_t)HKV * D * C * 2;
  bf16* wcb = (bf16*)w; w += (size_t)C * C * 2;
  bf16* qb  = (bf16*)w; w += M * (size_t)H * D * 2;
  bf16* kb  = (bf16*)w; w += M * (size_t)HKV * D * 2;
  bf16* vb  = (bf16*)w; w += M * (size_t)HKV * D * 2;
  bf16* yb  = (bf16*)w; w += M * C * 2;

  auto cvt = [&](const float* src, bf16* dst, size_t n) {
    int n4 = (int)(n / 4);
    cvt_f32_bf16<<<dim3((n4 + 255) / 256), dim3(256), 0, stream>>>(src, dst, n4);
  };
  cvt(x,  xb,  M * C);
  cvt(Wq, wqb, (size_t)H * D * C);
  cvt(Wk, wkb, (size_t)HKV * D * C);
  cvt(Wv, wvb, (size_t)HKV * D * C);
  cvt(Wc, wcb, (size_t)C * C);

  // QKV projections (bf16 out), 128x128 block tiles
  gemm_xt_wmma<false><<<dim3(C / 128, M / 128), 128, 0, stream>>>(
      xb, wqb, qb, (int)M, C, C);
  gemm_xt_wmma<false><<<dim3((HKV * D) / 128, M / 128), 128, 0, stream>>>(
      xb, wkb, kb, (int)M, HKV * D, C);
  gemm_xt_wmma<false><<<dim3((HKV * D) / 128, M / 128), 128, 0, stream>>>(
      xb, wvb, vb, (int)M, HKV * D, C);

  // RoPE in place on Q and K
  {
    int totq = (int)(M * H * 32);
    rope_kernel<<<dim3((totq + 255) / 256), dim3(256), 0, stream>>>(
        qb, pos, (int)M, H, T);
    int totk = (int)(M * HKV * 32);
    rope_kernel<<<dim3((totk + 255) / 256), dim3(256), 0, stream>>>(
        kb, pos, (int)M, HKV, T);
  }

  // Causal flash attention (TDM-staged K, 64-key blocks)
  flash_attn<<<dim3(T / 64, H, B), dim3(128), 0, stream>>>(qb, kb, vb, yb);

  // Output projection (f32 out straight to d_out)
  gemm_xt_wmma<true><<<dim3(C / 128, M / 128), 128, 0, stream>>>(
      yb, wcb, d_out, (int)M, C, C);
}
